// Transformer_42090679501224
// MI455X (gfx1250) — matrix operand order repaired and checked by reading, not compiled, not run on hardware
//
#include <hip/hip_runtime.h>
#include <hip/hip_bf16.h>
#include <math.h>

// ---------------- problem constants (from reference) ----------------
#define VOCAB 32000
#define DMODEL 384
#define NHEAD 6
#define NLAYER 6
#define FFNH 1536
#define BATCH 38
#define SEQ 256
#define HD (DMODEL / NHEAD)           // 64
#define MROWS (BATCH * SEQ)           // 9728
#define QKVD (3 * DMODEL)             // 1152

typedef __attribute__((ext_vector_type(16))) _Float16 v16h;
typedef __attribute__((ext_vector_type(4)))  _Float16 v4h;
typedef __attribute__((ext_vector_type(8)))  float    v8f;
typedef __attribute__((ext_vector_type(4)))  float    v4f;
typedef __attribute__((ext_vector_type(8)))  unsigned int v8u;

// ---- A-fragment (16x32 f16) LDS load per CDNA5 wave32 layout ----
__device__ __forceinline__ v16h load_a_frag(const _Float16 (*T)[40], int row, int hi) {
  v8u u;
#pragma unroll
  for (int v = 0; v < 8; ++v) {
    int k = ((v < 4) ? (2 * v) : (16 + 2 * (v - 4))) + hi * 8;
    u[v] = *(const unsigned int*)&T[row][k];
  }
  return __builtin_bit_cast(v16h, u);
}
// ---- B-fragment (32x16 f16) LDS load ----
__device__ __forceinline__ v16h load_b_frag(const _Float16 (*T)[40], int row, int hi) {
  v8u u;
  (void)0;
#pragma unroll
  for (int v = 0; v < 8; ++v)
    u[v] = *(const unsigned int*)&T[row][((v < 4) ? (2 * v) : (16 + 2 * (v - 4))) + hi * 8];
  return __builtin_bit_cast(v16h, u);
}
__device__ __forceinline__ v4h cvt4(v4f v) {
  return (v4h){ (_Float16)v.x, (_Float16)v.y, (_Float16)v.z, (_Float16)v.w };
}

// =====================================================================
// NT GEMM:  C[m,n] = sum_k A[m,k] * B[n,k] (+ bias[n])
// 128x128x32 block, 256 threads = 8 waves (2M x 4N); each wave owns a
// 64x32 patch = 4x2 grid of 16x16 f32 WMMA accumulators.
// All dims exact multiples of the tile -> no bounds checks.
// Batched over blockIdx.z with (z/zdiv, z%zdiv) stride pairs.
// =====================================================================
__global__ __launch_bounds__(256)
void gemm_nt_kernel(int K,
                    const float* __restrict__ A, int lda, long long aS0, long long aS1,
                    const float* __restrict__ B, int ldb, long long bS0, long long bS1,
                    float* __restrict__ C, int ldc, long long cS0, long long cS1,
                    const float* __restrict__ bias, int zdiv) {
  const int z  = blockIdx.z;
  const int z0 = z / zdiv, z1 = z % zdiv;
  A += (size_t)(z0 * aS0 + z1 * aS1);
  B += (size_t)(z0 * bS0 + z1 * bS1);
  C += (size_t)(z0 * cS0 + z1 * cS1);

  const int n0 = blockIdx.x * 128;
  const int m0 = blockIdx.y * 128;

  __shared__ __align__(16) _Float16 As[128][40];   // 80B row stride
  __shared__ __align__(16) _Float16 Bs[128][40];

  const int tid   = threadIdx.x;
  const int lane  = tid & 31;
  const int wave  = tid >> 5;            // 0..7
  const int waveM = (wave >> 2) * 64;    // 0 or 64
  const int waveN = (wave & 3) * 32;     // 0,32,64,96
  const int mIdx  = lane & 15;
  const int hi    = lane >> 4;

  const int arow = tid >> 3;             // 0..31 (chunk row base)
  const int acol = (tid & 7) * 4;        // k offset 0..28

  v8f acc[4][2];
#pragma unroll
  for (int i = 0; i < 4; ++i)
#pragma unroll
    for (int j = 0; j < 2; ++j) acc[i][j] = (v8f){0,0,0,0,0,0,0,0};

  for (int k0 = 0; k0 < K; k0 += 32) {
    // ---- phase 1: issue all global loads (8x b128), no intervening waits ----
    v4f ta[4], tb[4];
#pragma unroll
    for (int i = 0; i < 4; ++i)
      ta[i] = *(const v4f*)(A + (size_t)(m0 + arow + i * 32) * lda + (k0 + acol));
#pragma unroll
    for (int i = 0; i < 4; ++i)
      tb[i] = *(const v4f*)(B + (size_t)(n0 + arow + i * 32) * ldb + (k0 + acol));
    // ---- phase 2: convert + LDS store ----
#pragma unroll
    for (int i = 0; i < 4; ++i) *(v4h*)&As[arow + i * 32][acol] = cvt4(ta[i]);
#pragma unroll
    for (int i = 0; i < 4; ++i) *(v4h*)&Bs[arow + i * 32][acol] = cvt4(tb[i]);
    __syncthreads();

    // prefetch next k tile into caches while this one computes
    if (k0 + 32 < K) {
      __builtin_prefetch(A + (size_t)(m0 + arow) * lda + (k0 + 32 + acol), 0, 0);
      __builtin_prefetch(B + (size_t)(n0 + arow) * ldb + (k0 + 32 + acol), 0, 0);
    }

    // ---- fragments + WMMA (4x2 per wave) ----
    v16h aF[4], bF[2];
#pragma unroll
    for (int t = 0; t < 4; ++t) aF[t] = load_a_frag(As, waveM + t * 16 + mIdx, hi);
#pragma unroll
    for (int t = 0; t < 2; ++t) bF[t] = load_b_frag(Bs, waveN + t * 16 + mIdx, hi);
#pragma unroll
    for (int tm = 0; tm < 4; ++tm)
#pragma unroll
      for (int tn = 0; tn < 2; ++tn)
        acc[tm][tn] = __builtin_amdgcn_wmma_f32_16x16x32_f16(
            false, aF[tm], false, bF[tn], (short)0, acc[tm][tn], false, false);
    __syncthreads();
  }

  // ---- store C (+bias); C/D layout: VGPR v -> m = v + 8*hi, n = lane&15 ----
#pragma unroll
  for (int tm = 0; tm < 4; ++tm)
#pragma unroll
    for (int tn = 0; tn < 2; ++tn) {
      int mBase = m0 + waveM + tm * 16 + hi * 8;
      int n     = n0 + waveN + tn * 16 + mIdx;
      float bv = bias ? bias[n] : 0.0f;
#pragma unroll
      for (int v = 0; v < 8; ++v)
        C[(size_t)(mBase + v) * ldc + n] = acc[tm][tn][v] + bv;
    }
}

// =====================================================================
// NN GEMM (attn @ V): C[m,n] = sum_k A[m,k] * B[k,n]
// 64x64x32 block, 128 threads = 4 waves (2M x 2N), 2x2 accum per wave.
// =====================================================================
__global__ __launch_bounds__(128)
void gemm_nn_kernel(int K,
                    const float* __restrict__ A, int lda, long long aS0, long long aS1,
                    const float* __restrict__ B, int ldb, long long bS0, long long bS1,
                    float* __restrict__ C, int ldc, long long cS0, long long cS1,
                    int zdiv) {
  const int z  = blockIdx.z;
  const int z0 = z / zdiv, z1 = z % zdiv;
  A += (size_t)(z0 * aS0 + z1 * aS1);
  B += (size_t)(z0 * bS0 + z1 * bS1);
  C += (size_t)(z0 * cS0 + z1 * cS1);

  const int n0 = blockIdx.x * 64;
  const int m0 = blockIdx.y * 64;

  __shared__ __align__(16) _Float16 As[64][40];
  __shared__ __align__(16) _Float16 Bs[64][40];   // [n][k]

  const int tid   = threadIdx.x;
  const int lane  = tid & 31;
  const int wave  = tid >> 5;
  const int waveM = (wave >> 1) * 32;
  const int waveN = (wave & 1) * 32;
  const int mIdx  = lane & 15;
  const int hi    = lane >> 4;

  const int arow  = tid >> 3;            // 0..15
  const int acol  = (tid & 7) * 4;
  const int bkrow = tid >> 4;            // 0..7
  const int bncol = (tid & 15) * 4;

  v8f acc[2][2];
#pragma unroll
  for (int i = 0; i < 2; ++i)
#pragma unroll
    for (int j = 0; j < 2; ++j) acc[i][j] = (v8f){0,0,0,0,0,0,0,0};

  for (int k0 = 0; k0 < K; k0 += 32) {
    // phase 1: all global loads
    v4f ta[4], tb[4];
#pragma unroll
    for (int i = 0; i < 4; ++i)
      ta[i] = *(const v4f*)(A + (size_t)(m0 + arow + i * 16) * lda + (k0 + acol));
#pragma unroll
    for (int i = 0; i < 4; ++i)
      tb[i] = *(const v4f*)(B + (size_t)(k0 + bkrow + i * 8) * ldb + (n0 + bncol));
    // phase 2: convert + LDS store (B transposed into k-major rows)
#pragma unroll
    for (int i = 0; i < 4; ++i) *(v4h*)&As[arow + i * 16][acol] = cvt4(ta[i]);
#pragma unroll
    for (int i = 0; i < 4; ++i) {
      int k = bkrow + i * 8;
      Bs[bncol + 0][k] = (_Float16)tb[i].x;
      Bs[bncol + 1][k] = (_Float16)tb[i].y;
      Bs[bncol + 2][k] = (_Float16)tb[i].z;
      Bs[bncol + 3][k] = (_Float16)tb[i].w;
    }
    __syncthreads();

    if (k0 + 32 < K) {
      __builtin_prefetch(A + (size_t)(m0 + arow) * lda + (k0 + 32 + acol), 0, 0);
      __builtin_prefetch(B + (size_t)(k0 + 32 + bkrow) * ldb + (n0 + bncol), 0, 0);
    }

    v16h aF[2], bF[2];
#pragma unroll
    for (int t = 0; t < 2; ++t) aF[t] = load_a_frag(As, waveM + t * 16 + mIdx, hi);
#pragma unroll
    for (int t = 0; t < 2; ++t) bF[t] = load_b_frag(Bs, waveN + t * 16 + mIdx, hi);
#pragma unroll
    for (int tm = 0; tm < 2; ++tm)
#pragma unroll
      for (int tn = 0; tn < 2; ++tn)
        acc[tm][tn] = __builtin_amdgcn_wmma_f32_16x16x32_f16(
            false, aF[tm], false, bF[tn], (short)0, acc[tm][tn], false, false);
    __syncthreads();
  }

#pragma unroll
  for (int tm = 0; tm < 2; ++tm)
#pragma unroll
    for (int tn = 0; tn < 2; ++tn) {
      int mBase = m0 + waveM + tm * 16 + hi * 8;
      int n     = n0 + waveN + tn * 16 + mIdx;
#pragma unroll
      for (int v = 0; v < 8; ++v)
        C[(size_t)(mBase + v) * ldc + n] = acc[tm][tn][v];
    }
}

// =====================================================================
// mask length: pad = (x != 0); row with max non-pad count (first on tie),
// max_len = first zero in that row (or SEQ). Single block.
// =====================================================================
__global__ void maxlen_kernel(const int* __restrict__ x, int* __restrict__ out) {
  __shared__ int cnt[BATCH];
  int t = threadIdx.x;
  if (t < BATCH) {
    int c = 0;
    for (int s = 0; s < SEQ; ++s) c += (x[t * SEQ + s] != 0) ? 1 : 0;
    cnt[t] = c;
  }
  __syncthreads();
  if (t == 0) {
    int best = 0;
    for (int r = 1; r < BATCH; ++r)
      if (cnt[r] > cnt[best]) best = r;
    int ml = SEQ;
    for (int j = 0; j < SEQ; ++j)
      if (x[best * SEQ + j] == 0) { ml = j; break; }
    *out = ml;
  }
}

// ===================================================================== embedding + PE
__global__ void embed_kernel(const int* __restrict__ x, const float* __restrict__ emb,
                             float* __restrict__ h) {
  int row = blockIdx.x;
  int s   = row % SEQ;
  int tok = x[row];
  const float scale = sqrtf((float)DMODEL);
  for (int d = threadIdx.x; d < DMODEL; d += blockDim.x) {
    int even = d & ~1;
    float denom = powf(10000.0f, (float)even / (float)DMODEL);
    float ang = (float)s / denom;
    float pe = (d & 1) ? cosf(ang) : sinf(ang);
    h[(size_t)row * DMODEL + d] = emb[(size_t)tok * DMODEL + d] * scale + pe;
  }
}

// ===================================================================== masked softmax
__global__ __launch_bounds__(SEQ)
void softmax_kernel(float* __restrict__ scores, const int* __restrict__ maxlen_p,
                    float scale) {
  __shared__ float red[SEQ];
  const int i = blockIdx.x;
  const int z = blockIdx.y;
  const int j = threadIdx.x;
  const int ml = *maxlen_p;
  float* row = scores + (size_t)z * SEQ * SEQ + (size_t)i * SEQ;

  bool allowed = (j <= i) && !((i >= ml) && (j >= ml));
  float v = allowed ? row[j] * scale : -INFINITY;

  red[j] = v; __syncthreads();
  for (int off = SEQ / 2; off > 0; off >>= 1) {
    if (j < off) red[j] = fmaxf(red[j], red[j + off]);
    __syncthreads();
  }
  float m = red[0]; __syncthreads();

  float e = allowed ? expf(v - m) : 0.0f;
  red[j] = e; __syncthreads();
  for (int off = SEQ / 2; off > 0; off >>= 1) {
    if (j < off) red[j] += red[j + off];
    __syncthreads();
  }
  row[j] = e / red[0];
}

// ===================================================================== residual + LN
__global__ __launch_bounds__(128)
void ln_residual_kernel(float* __restrict__ h, const float* __restrict__ delta,
                        const float* __restrict__ gamma, const float* __restrict__ beta) {
  __shared__ float s1[128], s2[128];
  const int row = blockIdx.x;
  const int t = threadIdx.x;
  float* hr = h + (size_t)row * DMODEL;
  const float* dr = delta + (size_t)row * DMODEL;

  float vals[3];
  float sum = 0.0f, sumsq = 0.0f;
#pragma unroll
  for (int i = 0; i < 3; ++i) {
    int d = t + i * 128;
    float v = hr[d] + dr[d];
    vals[i] = v; sum += v; sumsq += v * v;
  }
  s1[t] = sum; s2[t] = sumsq; __syncthreads();
  for (int off = 64; off > 0; off >>= 1) {
    if (t < off) { s1[t] += s1[t + off]; s2[t] += s2[t + off]; }
    __syncthreads();
  }
  float mean = s1[0] / (float)DMODEL;
  float var  = s2[0] / (float)DMODEL - mean * mean;
  float inv  = rsqrtf(var + 1e-5f);
#pragma unroll
  for (int i = 0; i < 3; ++i) {
    int d = t + i * 128;
    hr[d] = gamma[d] * (vals[i] - mean) * inv + beta[d];
  }
}

// ===================================================================== exact GeLU
__global__ void gelu_kernel(float* __restrict__ p, long long n) {
  long long i = (long long)blockIdx.x * blockDim.x + threadIdx.x;
  long long stride = (long long)gridDim.x * blockDim.x;
  for (; i < n; i += stride) {
    float x = p[i];
    p[i] = 0.5f * x * (1.0f + erff(x * 0.70710678118654752f));
  }
}

// ===================================================================== NLL per row
__global__ __launch_bounds__(256)
void nll_kernel(const float* __restrict__ logits, const int* __restrict__ targets,
                float* __restrict__ nll) {
  __shared__ float red[256];
  const int row = blockIdx.x;
  const int t = threadIdx.x;
  const float* lr = logits + (size_t)row * VOCAB;

  float m = -INFINITY;
  for (int c = t; c < VOCAB; c += 256) m = fmaxf(m, lr[c]);
  red[t] = m; __syncthreads();
  for (int off = 128; off > 0; off >>= 1) {
    if (t < off) red[t] = fmaxf(red[t], red[t + off]);
    __syncthreads();
  }
  float rowmax = red[0]; __syncthreads();

  float s = 0.0f;
  for (int c = t; c < VOCAB; c += 256) s += expf(lr[c] - rowmax);
  red[t] = s; __syncthreads();
  for (int off = 128; off > 0; off >>= 1) {
    if (t < off) red[t] += red[t + off];
    __syncthreads();
  }
  if (t == 0) {
    int tgt = targets[row];
    nll[row] = -(lr[tgt] - rowmax - logf(red[0]));
  }
}

__global__ __launch_bounds__(256)
void loss_reduce_kernel(const float* __restrict__ nll, float* __restrict__ loss_out) {
  __shared__ float red[256];
  int t = threadIdx.x;
  float s = 0.0f;
  for (int i = t; i < MROWS; i += 256) s += nll[i];
  red[t] = s; __syncthreads();
  for (int off = 128; off > 0; off >>= 1) {
    if (t < off) red[t] += red[t + off];
    __syncthreads();
  }
  if (t == 0) *loss_out = red[0] / (float)MROWS;
}

// =====================================================================
// host-side orchestration
// =====================================================================
static inline void launch_gemm_nt(int M, int N, int K,
                                  const float* A, int lda, long long aS0, long long aS1,
                                  const float* B, int ldb, long long bS0, long long bS1,
                                  float* C, int ldc, long long cS0, long long cS1,
                                  const float* bias, int Z, int zdiv, hipStream_t stream) {
  dim3 grid(N / 128, M / 128, Z);
  gemm_nt_kernel<<<grid, 256, 0, stream>>>(K, A, lda, aS0, aS1, B, ldb, bS0, bS1,
                                           C, ldc, cS0, cS1, bias, zdiv);
}
static inline void launch_gemm_nn(int M, int N, int K,
                                  const float* A, int lda, long long aS0, long long aS1,
                                  const float* B, int ldb, long long bS0, long long bS1,
                                  float* C, int ldc, long long cS0, long long cS1,
                                  int Z, int zdiv, hipStream_t stream) {
  dim3 grid(N / 64, M / 64, Z);
  gemm_nn_kernel<<<grid, 128, 0, stream>>>(K, A, lda, aS0, aS1, B, ldb, bS0, bS1,
                                           C, ldc, cS0, cS1, zdiv);
}

extern "C" void kernel_launch(void* const* d_in, const int* in_sizes, int n_in,
                              void* d_out, int out_size, void* d_ws, size_t ws_size,
                              hipStream_t stream) {
  (void)in_sizes; (void)n_in; (void)out_size; (void)ws_size;
  const int*   x       = (const int*)  d_in[0];
  const int*   targets = (const int*)  d_in[1];
  const float* emb     = (const float*)d_in[2];
  const float* Wqkv    = (const float*)d_in[3];
  const float* bqkv    = (const float*)d_in[4];
  const float* Wo      = (const float*)d_in[5];
  const float* bo      = (const float*)d_in[6];
  const float* gamma1  = (const float*)d_in[7];
  const float* beta1   = (const float*)d_in[8];
  const float* W1      = (const float*)d_in[9];
  const float* b1      = (const float*)d_in[10];
  const float* W2      = (const float*)d_in[11];
  const float* b2      = (const float*)d_in[12];
  const float* gamma2  = (const float*)d_in[13];
  const float* beta2   = (const float*)d_in[14];
  const float* Wout    = (const float*)d_in[15];
  const float* bout    = (const float*)d_in[16];

  float* logits = (float*)d_out;                                   // (B,S,V)
  float* loss   = logits + (size_t)MROWS * VOCAB;                  // scalar

  // ---- workspace layout (floats) ----
  float* ws = (float*)d_ws;
  size_t off = 0;
  float* h      = ws + off; off += (size_t)MROWS * DMODEL;
  float* tmp    = ws + off; off += (size_t)MROWS * DMODEL;
  float* qkv    = ws + off; off += (size_t)MROWS * QKVD;
  float* scores = ws + off; off += (size_t)BATCH * NHEAD * SEQ * SEQ;
  float* vals   = ws + off; off += (size_t)MROWS * DMODEL;
  float* ffh    = ws + off; off += (size_t)MROWS * FFNH;
  float* nll    = ws + off; off += (size_t)MROWS;
  int*   maxlen = (int*)(ws + off); off += 16;

  const float inv_sqrt_dk = 1.0f / sqrtf((float)HD);

  // 1) data-dependent mask length + embedding
  maxlen_kernel<<<1, 256, 0, stream>>>(x, maxlen);
  embed_kernel<<<MROWS, 128, 0, stream>>>(x, emb, h);

  // 2) transformer layers
  for (int l = 0; l < NLAYER; ++l) {
    const float* Wq_l = Wqkv + (size_t)l * QKVD * DMODEL;
    const float* bq_l = bqkv + (size_t)l * QKVD;
    const float* Wo_l = Wo   + (size_t)l * DMODEL * DMODEL;
    const float* bo_l = bo   + (size_t)l * DMODEL;
    const float* g1_l = gamma1 + (size_t)l * DMODEL;
    const float* be1_l = beta1 + (size_t)l * DMODEL;
    const float* W1_l = W1   + (size_t)l * FFNH * DMODEL;
    const float* b1_l = b1   + (size_t)l * FFNH;
    const float* W2_l = W2   + (size_t)l * DMODEL * FFNH;
    const float* b2_l = b2   + (size_t)l * DMODEL;
    const float* g2_l = gamma2 + (size_t)l * DMODEL;
    const float* be2_l = beta2 + (size_t)l * DMODEL;

    // qkv = h @ Wqkv^T + bqkv      (9728 x 1152, K=384)
    launch_gemm_nt(MROWS, QKVD, DMODEL, h, DMODEL, 0, 0,
                   Wq_l, DMODEL, 0, 0, qkv, QKVD, 0, 0, bq_l, 1, 1, stream);

    // scores[z] = q_z @ k_z^T, z = b*NHEAD + h  (256 x 256, K=64)
    launch_gemm_nt(SEQ, SEQ, HD,
                   qkv + 0,  QKVD, (long long)SEQ * QKVD, 3 * HD,       // q
                   qkv + HD, QKVD, (long long)SEQ * QKVD, 3 * HD,       // k
                   scores, SEQ,
                   (long long)NHEAD * SEQ * SEQ, (long long)SEQ * SEQ,
                   nullptr, BATCH * NHEAD, NHEAD, stream);

    // softmax with mask + scale
    {
      dim3 g(SEQ, BATCH * NHEAD);
      softmax_kernel<<<g, SEQ, 0, stream>>>(scores, maxlen, inv_sqrt_dk);
    }

    // vals[z] = attn_z @ v_z   (NN: 256 x 64, K=256)
    launch_gemm_nn(SEQ, HD, SEQ,
                   scores, SEQ, (long long)NHEAD * SEQ * SEQ, (long long)SEQ * SEQ,
                   qkv + 2 * HD, QKVD, (long long)SEQ * QKVD, 3 * HD,   // v
                   vals, DMODEL, (long long)SEQ * DMODEL, HD,
                   BATCH * NHEAD, NHEAD, stream);

    // att_out = vals @ Wo^T + bo
    launch_gemm_nt(MROWS, DMODEL, DMODEL, vals, DMODEL, 0, 0,
                   Wo_l, DMODEL, 0, 0, tmp, DMODEL, 0, 0, bo_l, 1, 1, stream);

    // h = LN(h + att_out)
    ln_residual_kernel<<<MROWS, 128, 0, stream>>>(h, tmp, g1_l, be1_l);

    // ffh = gelu(h @ W1^T + b1)
    launch_gemm_nt(MROWS, FFNH, DMODEL, h, DMODEL, 0, 0,
                   W1_l, DMODEL, 0, 0, ffh, FFNH, 0, 0, b1_l, 1, 1, stream);
    gelu_kernel<<<2048, 256, 0, stream>>>(ffh, (long long)MROWS * FFNH);

    // ffn_out = ffh @ W2^T + b2
    launch_gemm_nt(MROWS, DMODEL, FFNH, ffh, FFNH, 0, 0,
                   W2_l, FFNH, 0, 0, tmp, DMODEL, 0, 0, b2_l, 1, 1, stream);

    // h = LN(h + ffn_out)
    ln_residual_kernel<<<MROWS, 128, 0, stream>>>(h, tmp, g2_l, be2_l);
  }

  // 3) logits = h @ Wout^T + bout   (9728 x 32000, K=384) -> d_out
  launch_gemm_nt(MROWS, VOCAB, DMODEL, h, DMODEL, 0, 0,
                 Wout, DMODEL, 0, 0, logits, VOCAB, 0, 0, bout, 1, 1, stream);

  // 4) loss = mean NLL
  nll_kernel<<<MROWS, 256, 0, stream>>>(logits, targets, nll);
  loss_reduce_kernel<<<1, 256, 0, stream>>>(nll, loss);
}
